// AttnBlockpp3d_old_70463233458259
// MI455X (gfx1250) — matrix operand order repaired and checked by reading, not compiled, not run on hardware
//
#include <hip/hip_runtime.h>

typedef _Float16 v16h __attribute__((ext_vector_type(16)));
typedef _Float16 h8   __attribute__((ext_vector_type(8)));
typedef float    v8f  __attribute__((ext_vector_type(8)));

#define B_SZ   16
#define C_SZ   256
#define HW     1024
#define HEADS  4
#define CH     64
#define EPSV   1e-6f

// ---- WMMA fragment loaders (contiguous k order, same in A and B) ----
// A 16x32 f16: lane l (l15=l&15, hi=l>>4): row M=l15; element e -> K = 16*hi + e (contiguous run of 16)
static __device__ __forceinline__ v16h loadA(const _Float16* rowp, int kbase, int hi) {
  h8 lo = *(const h8*)(rowp + kbase + 16 * hi);
  h8 hh = *(const h8*)(rowp + kbase + 16 * hi + 8);
  v16h a;
#pragma unroll
  for (int i = 0; i < 8; ++i) { a[i] = lo[i]; a[i + 8] = hh[i]; }
  return a;
}
// B 32x16 f16: lane l: col N=l15; element e -> K = 16*hi + e (contiguous run of 16)
static __device__ __forceinline__ v16h loadB(const _Float16* p) {
  h8 lo = *(const h8*)(p);
  h8 hh = *(const h8*)(p + 8);
  v16h b;
#pragma unroll
  for (int i = 0; i < 8; ++i) { b[i] = lo[i]; b[i + 8] = hh[i]; }
  return b;
}

// ---- GroupNorm: one block per (batch, group); 8 ch x 1024 px = 8192 values ----
__global__ void k_groupnorm(const float* __restrict__ x,
                            const float* __restrict__ scale,
                            const float* __restrict__ bias,
                            _Float16* __restrict__ h) {
  int bg = blockIdx.x;            // 0..511
  int b = bg >> 5, g = bg & 31;
  const float* xin = x + (size_t)(b * C_SZ + g * 8) * HW;
  __shared__ float ssum[256], ssq[256];
  float s = 0.f, sq = 0.f;
  for (int i = threadIdx.x; i < 8 * HW; i += 256) {
    float v = xin[i];
    s += v; sq += v * v;
  }
  ssum[threadIdx.x] = s; ssq[threadIdx.x] = sq;
  __syncthreads();
  for (int off = 128; off > 0; off >>= 1) {
    if ((int)threadIdx.x < off) {
      ssum[threadIdx.x] += ssum[threadIdx.x + off];
      ssq[threadIdx.x]  += ssq[threadIdx.x + off];
    }
    __syncthreads();
  }
  float mean = ssum[0] * (1.f / 8192.f);
  float var  = ssq[0] * (1.f / 8192.f) - mean * mean;
  float inv  = rsqrtf(var + EPSV);
  for (int i = threadIdx.x; i < 8 * HW; i += 256) {
    int cl = i >> 10;
    int pos = i & 1023;
    int c = g * 8 + cl;
    float v = (xin[i] - mean) * inv * scale[c] + bias[c];
    h[(size_t)(b * HW + pos) * C_SZ + c] = (_Float16)v;   // [pos][C] layout
  }
}

// ---- Convert + transpose weights to f16: wt[w][d*256 + c] = W[w][c*256 + d] ----
__global__ void k_prep_w(const float* __restrict__ W0, const float* __restrict__ W1,
                         const float* __restrict__ W2, const float* __restrict__ W3,
                         _Float16* __restrict__ wt) {
  int i = blockIdx.x * 256 + threadIdx.x;   // 4 * 65536 total
  int w = i >> 16; int e = i & 65535;
  int c = e >> 8, d = e & 255;
  const float* W = (w == 0) ? W0 : (w == 1) ? W1 : (w == 2) ? W2 : W3;
  wt[(size_t)w * 65536 + (size_t)d * 256 + c] = (_Float16)W[(size_t)c * 256 + d];
}

// ---- q/k/v projections: one wave per 16x16 output tile, WMMA over K=256 ----
__global__ void k_proj(const _Float16* __restrict__ h,
                       const _Float16* __restrict__ wt,
                       const float* __restrict__ b0,
                       const float* __restrict__ b1,
                       const float* __restrict__ b2,
                       _Float16* __restrict__ Qb,   // [hb][pos][64]  (prescaled 1/8)
                       _Float16* __restrict__ Kb,   // [hb][pos][64]
                       _Float16* __restrict__ Vt) { // [hb][64][pos]
  int lane = threadIdx.x & 31;
  int l15 = lane & 15, hi = lane >> 4;
  int m0 = blockIdx.x * 16;
  int nt = blockIdx.y;            // 0..47 : 3 weights x 16 col-tiles
  int wsel = nt >> 4;
  int n0 = (nt & 15) * 16;
  const _Float16* A  = h + (size_t)(m0 + l15) * C_SZ;
  const _Float16* Bw = wt + (size_t)wsel * 65536 + (size_t)(n0 + l15) * C_SZ;
  v8f acc = {};
#pragma unroll
  for (int k = 0; k < C_SZ; k += 32) {
    v16h a = loadA(A, k, hi);
    v16h b = loadB(Bw + k + 16 * hi);
    acc = __builtin_amdgcn_wmma_f32_16x16x32_f16(false, a, false, b, (short)0, acc, false, false);
  }
  const float* bias = (wsel == 0) ? b0 : (wsel == 1) ? b1 : b2;
  int col = n0 + l15;
  int head = col >> 6, cl = col & 63;
  float bv = bias[col];
#pragma unroll
  for (int r = 0; r < 8; ++r) {
    int pos = m0 + r + 8 * hi;
    int bb = pos >> 10, phw = pos & 1023;
    int hb = bb * HEADS + head;
    float v = acc[r] + bv;
    if (wsel == 0)
      Qb[(size_t)hb * HW * CH + (size_t)phw * CH + cl] = (_Float16)(v * 0.125f);
    else if (wsel == 1)
      Kb[(size_t)hb * HW * CH + (size_t)phw * CH + cl] = (_Float16)v;
    else
      Vt[(size_t)hb * HW * CH + (size_t)cl * HW + phw] = (_Float16)v;
  }
}

// ---- Flash attention: 1 wave = 16 queries of one (batch,head); online softmax ----
__global__ void k_attn(const _Float16* __restrict__ Qb,
                       const _Float16* __restrict__ Kb,
                       const _Float16* __restrict__ Vt,
                       _Float16* __restrict__ hh) {   // [pos][C] f16
  int lane = threadIdx.x & 31;
  int l15 = lane & 15, hi = lane >> 4;
  int q0 = blockIdx.x * 16;
  int hb = blockIdx.y;
  int bb = hb >> 2, head = hb & 3;
  const _Float16* Q = Qb + (size_t)hb * HW * CH;
  const _Float16* K = Kb + (size_t)hb * HW * CH;
  const _Float16* V = Vt + (size_t)hb * HW * CH;

  __shared__ _Float16 pL[16 * 32];   // P tile staging (D layout -> A layout)

  const _Float16* qrow = Q + (size_t)(q0 + l15) * CH;
  v16h qf0 = loadA(qrow, 0, hi);     // channels 0..31
  v16h qf1 = loadA(qrow, 32, hi);    // channels 32..63

  v8f zero = {};
  v8f accO[4];
  float mrow[8], lrow[8];
#pragma unroll
  for (int t = 0; t < 4; ++t) accO[t] = zero;
#pragma unroll
  for (int r = 0; r < 8; ++r) { mrow[r] = -1e30f; lrow[r] = 0.f; }

  for (int j = 0; j < HW; j += 32) {
    // S tile 16x32 = two 16x16 WMMA results, contraction over 64 channels
    v8f s0 = {}, s1 = {};
    {
      v16h bk = loadB(K + (size_t)(j + l15) * CH + 16 * hi);
      s0 = __builtin_amdgcn_wmma_f32_16x16x32_f16(false, qf0, false, bk, (short)0, s0, false, false);
      bk = loadB(K + (size_t)(j + l15) * CH + 32 + 16 * hi);
      s0 = __builtin_amdgcn_wmma_f32_16x16x32_f16(false, qf1, false, bk, (short)0, s0, false, false);
      bk = loadB(K + (size_t)(j + 16 + l15) * CH + 16 * hi);
      s1 = __builtin_amdgcn_wmma_f32_16x16x32_f16(false, qf0, false, bk, (short)0, s1, false, false);
      bk = loadB(K + (size_t)(j + 16 + l15) * CH + 32 + 16 * hi);
      s1 = __builtin_amdgcn_wmma_f32_16x16x32_f16(false, qf1, false, bk, (short)0, s1, false, false);
    }
    // Online softmax. D layout: acc element r <-> row (r + 8*hi), col l15.
    float p0a[8], p1a[8];
#pragma unroll
    for (int r = 0; r < 8; ++r) {
      float rm = fmaxf(s0[r], s1[r]);
      rm = fmaxf(rm, __shfl_xor(rm, 1));
      rm = fmaxf(rm, __shfl_xor(rm, 2));
      rm = fmaxf(rm, __shfl_xor(rm, 4));
      rm = fmaxf(rm, __shfl_xor(rm, 8));
      float mOld = mrow[r];
      float mNew = fmaxf(mOld, rm);
      float corr = __expf(mOld - mNew);
      float p0 = __expf(s0[r] - mNew);
      float p1 = __expf(s1[r] - mNew);
      float rs = p0 + p1;
      rs += __shfl_xor(rs, 1);
      rs += __shfl_xor(rs, 2);
      rs += __shfl_xor(rs, 4);
      rs += __shfl_xor(rs, 8);
      lrow[r] = lrow[r] * corr + rs;
      mrow[r] = mNew;
      p0a[r] = p0; p1a[r] = p1;
#pragma unroll
      for (int t = 0; t < 4; ++t) accO[t][r] *= corr;
    }
    __syncthreads();
#pragma unroll
    for (int r = 0; r < 8; ++r) {
      pL[(r + 8 * hi) * 32 + l15]      = (_Float16)p0a[r];
      pL[(r + 8 * hi) * 32 + 16 + l15] = (_Float16)p1a[r];
    }
    __syncthreads();
    // O += P (16x32) * V (32x64): 4 WMMA over c-tiles
    v16h pf = loadA(&pL[l15 * 32], 0, hi);
#pragma unroll
    for (int t = 0; t < 4; ++t) {
      v16h bv = loadB(V + (size_t)(t * 16 + l15) * HW + j + 16 * hi);
      accO[t] = __builtin_amdgcn_wmma_f32_16x16x32_f16(false, pf, false, bv, (short)0, accO[t], false, false);
    }
    __syncthreads();
  }
  // Normalize and store to hh f16 [pos][C]
#pragma unroll
  for (int r = 0; r < 8; ++r) {
    float inv = 1.f / lrow[r];
    int pos = q0 + r + 8 * hi;
#pragma unroll
    for (int t = 0; t < 4; ++t) {
      int ch = head * 64 + t * 16 + l15;
      hh[(size_t)(bb * HW + pos) * C_SZ + ch] = (_Float16)(accO[t][r] * inv);
    }
  }
}

// ---- Output projection + bias + residual: out = x + hh @ W3 + b3 ----
__global__ void k_final(const _Float16* __restrict__ hh,
                        const _Float16* __restrict__ w3t,
                        const float* __restrict__ b3,
                        const float* __restrict__ x,
                        float* __restrict__ out) {
  int lane = threadIdx.x & 31;
  int l15 = lane & 15, hi = lane >> 4;
  int m0 = blockIdx.x * 16;
  int n0 = blockIdx.y * 16;
  const _Float16* A  = hh + (size_t)(m0 + l15) * C_SZ;
  const _Float16* Bw = w3t + (size_t)(n0 + l15) * C_SZ;
  v8f acc = {};
#pragma unroll
  for (int k = 0; k < C_SZ; k += 32) {
    v16h a = loadA(A, k, hi);
    v16h b = loadB(Bw + k + 16 * hi);
    acc = __builtin_amdgcn_wmma_f32_16x16x32_f16(false, a, false, b, (short)0, acc, false, false);
  }
  int ch = n0 + l15;
  float bv = b3[ch];
#pragma unroll
  for (int r = 0; r < 8; ++r) {
    int pos = m0 + r + 8 * hi;
    int bb = pos >> 10, phw = pos & 1023;
    size_t idx = (size_t)(bb * C_SZ + ch) * HW + phw;
    out[idx] = x[idx] + acc[r] + bv;
  }
}

extern "C" void kernel_launch(void* const* d_in, const int* in_sizes, int n_in,
                              void* d_out, int out_size, void* d_ws, size_t ws_size,
                              hipStream_t stream) {
  (void)in_sizes; (void)n_in; (void)out_size; (void)ws_size;
  const float* x  = (const float*)d_in[0];
  const float* gs = (const float*)d_in[1];
  const float* gb = (const float*)d_in[2];
  const float* W0 = (const float*)d_in[3];
  const float* b0 = (const float*)d_in[4];
  const float* W1 = (const float*)d_in[5];
  const float* b1 = (const float*)d_in[6];
  const float* W2 = (const float*)d_in[7];
  const float* b2 = (const float*)d_in[8];
  const float* W3 = (const float*)d_in[9];
  const float* b3 = (const float*)d_in[10];
  float* out = (float*)d_out;

  char* ws = (char*)d_ws;
  _Float16* hbuf = (_Float16*)(ws);                                  // 8 MB  [pos][C]
  _Float16* wt   = (_Float16*)(ws + ((size_t)8 << 20));              // 512 KB (4 weights, transposed)
  _Float16* Qb   = (_Float16*)(ws + ((size_t)8 << 20) + ((size_t)512 << 10));
  _Float16* Kb   = Qb + (size_t)64 * HW * CH;                        // 8 MB each
  _Float16* Vt   = Kb + (size_t)64 * HW * CH;
  _Float16* hh   = Vt + (size_t)64 * HW * CH;                        // 8 MB

  k_prep_w   <<<1024, 256, 0, stream>>>(W0, W1, W2, W3, wt);
  k_groupnorm<<<512, 256, 0, stream>>>(x, gs, gb, hbuf);
  k_proj     <<<dim3(1024, 48), 32, 0, stream>>>(hbuf, wt, b0, b1, b2, Qb, Kb, Vt);
  k_attn     <<<dim3(64, 64), 32, 0, stream>>>(Qb, Kb, Vt, hh);
  k_final    <<<dim3(1024, 16), 32, 0, stream>>>(hh, wt + (size_t)3 * 65536, b3, x, out);
}
